// RespiratoryDynamicsNetwork_41420664603167
// MI455X (gfx1250) — matrix operand order repaired and checked by reading, not compiled, not run on hardware
//
#include <hip/hip_runtime.h>
#include <hip/hip_bf16.h>

typedef __attribute__((ext_vector_type(16))) _Float16 v16h;
typedef __attribute__((ext_vector_type(8)))  _Float16 v8h;
typedef __attribute__((ext_vector_type(8)))  float    v8f;

#define BATCH 64
#define TFR   200
#define NPOS  (BATCH * TFR)   // 12800 GEMM columns
#define NAUD  48000
#define NBINS 24001           // NAUD/2 + 1
#define FN1   240             // NAUD = FN1 * FN2
#define FN2   200

static __device__ __forceinline__ float sigmoidf(float x) { return 1.f / (1.f + expf(-x)); }

static __device__ __forceinline__ void interp_frame(int j, int& i0, int& i1, float& w)
{
    float src = ((float)j + 0.5f) * ((float)TFR / (float)NAUD) - 0.5f;
    src = fminf(fmaxf(src, 0.f), (float)(TFR - 1));
    i0 = (int)src;                 // src >= 0 so trunc == floor
    i1 = min(i0 + 1, TFR - 1);
    w  = src - (float)i0;
}

// ---------------------------------------------------------------------------
// Init: twiddle tables (forward convention W^i = exp(-2*pi*i/N)), the 5x32
// turbulence filter shapes, and zeroing of the atomic softmax accumulator.
// ---------------------------------------------------------------------------
__global__ void init_tables_kernel(float2* __restrict__ tw1, float2* __restrict__ tw2,
                                   float2* __restrict__ twN, float* __restrict__ fs,
                                   float* __restrict__ avg_shape)
{
    const float TWO_PI = 6.28318530717958647692f;
    for (int i = threadIdx.x; i < FN1; i += blockDim.x) {
        float a = TWO_PI * (float)i / (float)FN1;
        tw1[i] = make_float2(cosf(a), -sinf(a));
    }
    for (int i = threadIdx.x; i < FN2; i += blockDim.x) {
        float a = TWO_PI * (float)i / (float)FN2;
        tw2[i] = make_float2(cosf(a), -sinf(a));
    }
    for (int i = threadIdx.x; i < NAUD; i += blockDim.x) {
        float a = TWO_PI * (float)i / (float)NAUD;
        twN[i] = make_float2(cosf(a), -sinf(a));
    }
    for (int i = threadIdx.x; i < 5 * 32; i += blockDim.x) {
        int c = i / 32, q = i % 32;
        float lin = (float)q / 31.f;
        float v;
        if      (c == 0) { float z = (lin - 0.2f) / 0.10f; v = expf(-0.5f * z * z); }
        else if (c == 1) { float z = (lin - 0.1f) / 0.10f; v = expf(-0.5f * z * z); }
        else if (c == 2) { float z = (lin - 0.4f) / 0.15f; v = expf(-0.5f * z * z); }
        else if (c == 3) { float z = (lin - 0.6f) / 0.10f; v = expf(-0.5f * z * z); }
        else             { float l2 = 0.1f + 0.9f * lin;   v = l2 * l2; }
        fs[i] = v;
    }
    for (int i = threadIdx.x; i < BATCH * 32; i += blockDim.x) avg_shape[i] = 0.f;
}

// ---------------------------------------------------------------------------
// x_in = concat(condition, is_voiced)  -> (B, 129, T)
// ---------------------------------------------------------------------------
__global__ void build_input_kernel(const float* __restrict__ cond,
                                   const float* __restrict__ f0,
                                   float* __restrict__ xin)
{
    int gi = blockIdx.x * blockDim.x + threadIdx.x;
    const int total = BATCH * 129 * TFR;
    if (gi >= total) return;
    int t = gi % TFR;
    int c = (gi / TFR) % 129;
    int b = gi / (TFR * 129);
    float v;
    if (c < 128) v = cond[((size_t)b * 128 + c) * TFR + t];
    else         v = (f0[(size_t)b * TFR + t] > 0.f) ? 1.f : 0.f;
    xin[gi] = v;
}

// ---------------------------------------------------------------------------
// Weight pack: (OC, IC*KW) f32 -> (Mpad, Kpad) f16, zero padded.
// ---------------------------------------------------------------------------
__global__ void pack_weights_kernel(const float* __restrict__ w, _Float16* __restrict__ out,
                                    int OC, int Ktot, int Mpad, int Kpad)
{
    int gi = blockIdx.x * blockDim.x + threadIdx.x;
    if (gi >= Mpad * Kpad) return;
    int m = gi / Kpad, k = gi % Kpad;
    float v = (m < OC && k < Ktot) ? w[(size_t)m * Ktot + k] : 0.f;
    out[gi] = (_Float16)v;
}

// ---------------------------------------------------------------------------
// im2col pack: x (B, IC, T) f32 -> Bt (NPOS, Kpad) f16, K contiguous per
// column, zero padded (both K padding and conv boundary padding).
// grid = (ceil(Kpad/256), NPOS)
// ---------------------------------------------------------------------------
__global__ void im2col_kernel(const float* __restrict__ x, _Float16* __restrict__ bt,
                              int IC, int KW, int pad, int Kpad)
{
    int k = blockIdx.x * blockDim.x + threadIdx.x;
    int n = blockIdx.y;
    if (k >= Kpad) return;
    int Ktot = IC * KW;
    int b = n / TFR, t = n % TFR;
    float v = 0.f;
    if (k < Ktot) {
        int ic = k / KW;
        int dk = k - ic * KW;
        int st = t + dk - pad;
        if (st >= 0 && st < TFR) v = x[((size_t)b * IC + ic) * TFR + st];
    }
    bt[(size_t)n * Kpad + k] = (_Float16)v;
}

// ---------------------------------------------------------------------------
// GEMM on WMMA: one wave per 16(M) x 16(N) tile of y = A(Mpad,Kpad) *
// Bt^T(Kpad,NPOS).  No masking / no division in the K loop: fragments are
// two b128 loads per matrix per chunk, laid out per CDNA5 ISA 7.12.2.
// ---------------------------------------------------------------------------
__global__ void gemm_wmma_kernel(const _Float16* __restrict__ A,   // (Mpad, Kpad)
                                 const _Float16* __restrict__ Bt,  // (NPOS, Kpad)
                                 const float* __restrict__ bias,
                                 float* __restrict__ y,            // (B, OC, T)
                                 int Kpad, int OC, int leaky,
                                 int Mtiles, int Ntiles)
{
    int lane  = threadIdx.x & 31;
    int gwave = blockIdx.x * (blockDim.x >> 5) + (threadIdx.x >> 5);
    int total = Mtiles * Ntiles;
    if (gwave >= total) return;                 // wave-uniform exit (EXEC stays full)
    int mt = gwave % Mtiles;
    int nt = gwave / Mtiles;
    int m0 = mt * 16;
    int n0 = nt * 16;

    int lhi = (lane >= 16) ? 1 : 0;
    int mA  = m0 + (lane & 15);                 // A row this lane owns
    int nB  = n0 + (lane & 15);                 // B/C column this lane owns
    int b   = nB / TFR;
    int t   = nB % TFR;

    const _Float16* Arow = A  + (size_t)mA * Kpad;  // rows exist up to Mpad
    const _Float16* Brow = Bt + (size_t)nB * Kpad;

    __builtin_prefetch(Arow, 0, 0);
    __builtin_prefetch(Brow, 0, 0);

    v8f c;
#pragma unroll
    for (int r = 0; r < 8; ++r) c[r] = 0.f;

#pragma unroll 2
    for (int k0 = 0; k0 < Kpad; k0 += 32) {
        // A fragment: lane K-list = {k0+8*lhi+0..7, k0+16+8*lhi+0..7}
        v8h a_lo = *(const v8h*)(Arow + k0 + (lhi << 3));
        v8h a_hi = *(const v8h*)(Arow + k0 + 16 + (lhi << 3));
        // B fragment: lane K-list = {k0+8*lhi+0..7, k0+16+8*lhi+0..7}, N = lane&15
        v8h b_lo = *(const v8h*)(Brow + k0 + (lhi << 3));
        v8h b_hi = *(const v8h*)(Brow + k0 + 16 + (lhi << 3));
        v16h a  = __builtin_shufflevector(a_lo, a_hi, 0, 1, 2, 3, 4, 5, 6, 7,
                                          8, 9, 10, 11, 12, 13, 14, 15);
        v16h bf = __builtin_shufflevector(b_lo, b_hi, 0, 1, 2, 3, 4, 5, 6, 7,
                                          8, 9, 10, 11, 12, 13, 14, 15);
        c = __builtin_amdgcn_wmma_f32_16x16x32_f16(false, a, false, bf,
                                                   (short)0, c, false, false);
    }

    // C/D layout: M = r + 8*(lane>=16), N = lane&15
#pragma unroll
    for (int r = 0; r < 8; ++r) {
        int m = m0 + r + (lhi << 3);
        if (m < OC) {
            float v = c[r] + bias[m];
            if (leaky) v = (v >= 0.f) ? v : 0.1f * v;
            y[((size_t)b * OC + m) * TFR + t] = v;
        }
    }
}

// ---------------------------------------------------------------------------
// Activations: sigmoids -> breath_features + control; two stable softmaxes.
// Time-mean of spectral softmax accumulated atomically into avg_shape.
// ---------------------------------------------------------------------------
__global__ void activations_kernel(const float* __restrict__ p,     // (B,40,T)
                                   float* __restrict__ bf_out,      // (B,3,T)
                                   float* __restrict__ control,     // (B,8,T)
                                   float* __restrict__ avg_shape)   // (B,32)
{
    int gi = blockIdx.x * blockDim.x + threadIdx.x;
    if (gi >= BATCH * TFR) return;
    int b = gi / TFR, t = gi % TFR;
    const float* pb = p + (size_t)b * 40 * TFR + t;

    float inh = sigmoidf(pb[0 * TFR]);
    float exh = sigmoidf(pb[1 * TFR]);
    float pr  = sigmoidf(pb[2 * TFR]);

    bf_out[((size_t)b * 3 + 0) * TFR + t] = pr;    // pressure
    bf_out[((size_t)b * 3 + 1) * TFR + t] = inh;   // inhal
    bf_out[((size_t)b * 3 + 2) * TFR + t] = exh;   // exhal

    float* cb = control + (size_t)b * 8 * TFR;
    cb[0 * TFR + t] = inh;
    cb[1 * TFR + t] = exh;
    cb[2 * TFR + t] = pr;

    // spectral_shape softmax over channels 3..34, averaged over time
    float mx = -3.4e38f;
    for (int c = 0; c < 32; ++c) mx = fmaxf(mx, pb[(3 + c) * TFR]);
    float ev[32]; float s = 0.f;
    for (int c = 0; c < 32; ++c) { ev[c] = expf(pb[(3 + c) * TFR] - mx); s += ev[c]; }
    float inv = 1.f / (s * (float)TFR);
    for (int c = 0; c < 32; ++c) atomicAdd(&avg_shape[b * 32 + c], ev[c] * inv);

    // turbulence softmax over channels 35..39
    float m2 = -3.4e38f;
    for (int c = 0; c < 5; ++c) m2 = fmaxf(m2, pb[(35 + c) * TFR]);
    float tv[5]; float s2 = 0.f;
    for (int c = 0; c < 5; ++c) { tv[c] = expf(pb[(35 + c) * TFR] - m2); s2 += tv[c]; }
    for (int c = 0; c < 5; ++c) cb[(3 + c) * TFR + t] = tv[c] / s2;
}

// ---------------------------------------------------------------------------
// Four-step complex DFT, N = 240*200.  Phase 1: 240-pt DFTs over columns
// (stride FN2), fused with W_N^(n2*k1) twiddle.  conj!=0 => inverse kernel
// (unscaled); 1/N applied at real extraction.
// ---------------------------------------------------------------------------
__global__ void fft_phase1_kernel(const float2* __restrict__ in, float2* __restrict__ out,
                                  const float2* __restrict__ tw1,   // 240
                                  const float2* __restrict__ twN,   // 48000
                                  int conj)
{
    int b    = blockIdx.x / (FN2 / 8);
    int grp  = blockIdx.x % (FN2 / 8);
    int col0 = grp * 8;
    __shared__ float2 tile[8][FN1];
    const float2* inb = in + (size_t)b * NAUD;
    for (int i = threadIdx.x; i < 8 * FN1; i += blockDim.x) {
        int cI = i / FN1, n1 = i % FN1;
        tile[cI][n1] = inb[(size_t)n1 * FN2 + (col0 + cI)];
    }
    __syncthreads();
    float2* outb = out + (size_t)b * NAUD;
    for (int i = threadIdx.x; i < 8 * FN1; i += blockDim.x) {
        int cI = i / FN1, k1 = i % FN1;
        int n2 = col0 + cI;
        float accr = 0.f, acci = 0.f;
        int idx = 0;
        for (int n1 = 0; n1 < FN1; ++n1) {
            float2 v = tile[cI][n1];
            float2 w = tw1[idx];
            float wi = conj ? -w.y : w.y;
            accr += v.x * w.x - v.y * wi;
            acci += v.x * wi + v.y * w.x;
            idx += k1; if (idx >= FN1) idx -= FN1;
        }
        float2 tw = twN[n2 * k1];               // n2*k1 <= 199*239 < 48000
        float twi = conj ? -tw.y : tw.y;
        float2 r;
        r.x = accr * tw.x - acci * twi;
        r.y = accr * twi + acci * tw.x;
        outb[(size_t)k1 * FN2 + n2] = r;
    }
}

// Phase 2: 200-pt DFTs over rows; final index k = k1 + 240*k2.
__global__ void fft_phase2_kernel(const float2* __restrict__ in, float2* __restrict__ out,
                                  const float2* __restrict__ tw2,   // 200
                                  int conj)
{
    int b    = blockIdx.x / (FN1 / 8);
    int grp  = blockIdx.x % (FN1 / 8);
    int row0 = grp * 8;
    __shared__ float2 tile[8][FN2];
    const float2* inb = in + (size_t)b * NAUD;
    for (int i = threadIdx.x; i < 8 * FN2; i += blockDim.x) {
        int r = i / FN2, n2 = i % FN2;
        tile[r][n2] = inb[(size_t)(row0 + r) * FN2 + n2];
    }
    __syncthreads();
    float2* outb = out + (size_t)b * NAUD;
    for (int i = threadIdx.x; i < 8 * FN2; i += blockDim.x) {
        int r = i / FN2, k2 = i % FN2;
        int k1 = row0 + r;
        float accr = 0.f, acci = 0.f;
        int idx = 0;
        for (int n2 = 0; n2 < FN2; ++n2) {
            float2 v = tile[r][n2];
            float2 w = tw2[idx];
            float wi = conj ? -w.y : w.y;
            accr += v.x * w.x - v.y * wi;
            acci += v.x * wi + v.y * w.x;
            idx += k2; if (idx >= FN2) idx -= FN2;
        }
        outb[(size_t)k1 + (size_t)k2 * FN1] = make_float2(accr, acci);
    }
}

// ---------------------------------------------------------------------------
__global__ void complexify_kernel(const float* __restrict__ xr, float2* __restrict__ xc)
{
    int gi = blockIdx.x * blockDim.x + threadIdx.x;
    if (gi >= BATCH * NAUD) return;
    xc[gi] = make_float2(xr[gi], 0.f);
}

__global__ void extract_real_kernel(const float2* __restrict__ in, float* __restrict__ outr,
                                    float scale)
{
    int gi = blockIdx.x * blockDim.x + threadIdx.x;
    if (gi >= BATCH * NAUD) return;
    outr[gi] = in[gi].x * scale;
}

// spectrum *= band gain (real, Hermitian-symmetric): band of bin i is the
// largest band with floor(band*24001/32) <= i.
__global__ void apply_gains_kernel(float2* __restrict__ spec, const float* __restrict__ avg_shape)
{
    int gi = blockIdx.x * blockDim.x + threadIdx.x;
    if (gi >= BATCH * NAUD) return;
    int b = gi / NAUD, k = gi % NAUD;
    int kk = (k <= NAUD / 2) ? k : (NAUD - k);
    int band = (32 * (kk + 1) - 1) / NBINS;
    band = min(band, 31);
    float g = avg_shape[b * 32 + band];
    spec[gi].x *= g; spec[gi].y *= g;
}

// spectrum *= total_filter = sum_c loc_w[b,c] * interp32(fs[c], bin)
__global__ void apply_total_filter_kernel(float2* __restrict__ spec,
                                          const float* __restrict__ loc_w,
                                          const float* __restrict__ fs)
{
    int gi = blockIdx.x * blockDim.x + threadIdx.x;
    if (gi >= BATCH * NAUD) return;
    int b = gi / NAUD, k = gi % NAUD;
    int kk = (k <= NAUD / 2) ? k : (NAUD - k);
    float src = ((float)kk + 0.5f) * (32.f / (float)NBINS) - 0.5f;
    src = fminf(fmaxf(src, 0.f), 31.f);
    int i0 = (int)src;
    int i1 = min(i0 + 1, 31);
    float w = src - (float)i0;
    float tf = 0.f;
#pragma unroll
    for (int c = 0; c < 5; ++c) {
        float f = fs[c * 32 + i0] * (1.f - w) + fs[c * 32 + i1] * w;
        tf += loc_w[b * 5 + c] * f;
    }
    spec[gi].x *= tf; spec[gi].y *= tf;
}

// per-row std (ddof=1)
__global__ void std_reduce_kernel(const float* __restrict__ shaped, float* __restrict__ stdev)
{
    int b = blockIdx.x;
    __shared__ float ssum[256], ssq[256];
    float s = 0.f, q = 0.f;
    for (int i = threadIdx.x; i < NAUD; i += 256) {
        float v = shaped[(size_t)b * NAUD + i];
        s += v; q += v * v;
    }
    ssum[threadIdx.x] = s; ssq[threadIdx.x] = q;
    __syncthreads();
    for (int off = 128; off > 0; off >>= 1) {
        if (threadIdx.x < off) {
            ssum[threadIdx.x] += ssum[threadIdx.x + off];
            ssq[threadIdx.x]  += ssq[threadIdx.x + off];
        }
        __syncthreads();
    }
    if (threadIdx.x == 0) {
        float mean = ssum[0] / (float)NAUD;
        float var  = (ssq[0] - (float)NAUD * mean * mean) / (float)(NAUD - 1);
        stdev[b] = sqrtf(fmaxf(var, 1e-30f));
    }
}

// loc_w[b,c] = mean over audio samples of interpolated turbulence channel c
__global__ void locw_reduce_kernel(const float* __restrict__ control, float* __restrict__ loc_w)
{
    int b = blockIdx.x;
    const float* cb = control + (size_t)b * 8 * TFR;
    float acc[5] = {0.f, 0.f, 0.f, 0.f, 0.f};
    for (int j = threadIdx.x; j < NAUD; j += 256) {
        int i0, i1; float w;
        interp_frame(j, i0, i1, w);
#pragma unroll
        for (int c = 0; c < 5; ++c)
            acc[c] += cb[(3 + c) * TFR + i0] * (1.f - w) + cb[(3 + c) * TFR + i1] * w;
    }
    __shared__ float red[5][256];
    for (int c = 0; c < 5; ++c) red[c][threadIdx.x] = acc[c];
    __syncthreads();
    for (int off = 128; off > 0; off >>= 1) {
        if (threadIdx.x < off)
            for (int c = 0; c < 5; ++c) red[c][threadIdx.x] += red[c][threadIdx.x + off];
        __syncthreads();
    }
    if (threadIdx.x < 5) loc_w[b * 5 + threadIdx.x] = red[threadIdx.x][0] / (float)NAUD;
}

// breath = (shaped/std*0.1) * (inh*(1-voiced) + exh*pr), complexified for FFT2
__global__ void envelope_kernel(const float* __restrict__ shaped,
                                const float* __restrict__ stdev,
                                const float* __restrict__ control,
                                const float* __restrict__ xin,   // voiced = channel 128
                                float2* __restrict__ outc)
{
    int gi = blockIdx.x * blockDim.x + threadIdx.x;
    if (gi >= BATCH * NAUD) return;
    int b = gi / NAUD, j = gi % NAUD;
    int i0, i1; float w;
    interp_frame(j, i0, i1, w);
    const float* cb = control + (size_t)b * 8 * TFR;
    float inh = cb[0 * TFR + i0] * (1.f - w) + cb[0 * TFR + i1] * w;
    float exh = cb[1 * TFR + i0] * (1.f - w) + cb[1 * TFR + i1] * w;
    float pr  = cb[2 * TFR + i0] * (1.f - w) + cb[2 * TFR + i1] * w;
    const float* vb = xin + ((size_t)b * 129 + 128) * TFR;
    float vo = vb[i0] * (1.f - w) + vb[i1] * w;
    float sn = shaped[gi] * (0.1f / stdev[b]);
    outc[gi] = make_float2(sn * inh * (1.f - vo) + sn * exh * pr, 0.f);
}

// ---------------------------------------------------------------------------
extern "C" void kernel_launch(void* const* d_in, const int* in_sizes, int n_in,
                              void* d_out, int out_size, void* d_ws, size_t ws_size,
                              hipStream_t stream)
{
    const float* cond = (const float*)d_in[0];
    const float* f0   = (const float*)d_in[1];
    const float* wn   = (const float*)d_in[2];
    const float* w1 = (const float*)d_in[3];
    const float* b1 = (const float*)d_in[4];
    const float* w2 = (const float*)d_in[5];
    const float* b2 = (const float*)d_in[6];
    const float* w3 = (const float*)d_in[7];
    const float* b3 = (const float*)d_in[8];
    const float* w4 = (const float*)d_in[9];
    const float* b4 = (const float*)d_in[10];
    float* out = (float*)d_out;
    float* bf_out = out + (size_t)BATCH * NAUD;   // breath_features after filtered audio

    char* base = (char*)d_ws;
    size_t off = 0;
    auto alloc = [&](size_t bytes) -> char* {
        char* p = base + off;
        off = (off + bytes + 255) & ~(size_t)255;
        return p;
    };
    float*     xin      = (float*)     alloc((size_t)BATCH * 129 * TFR * 4);
    float*     h1       = (float*)     alloc((size_t)BATCH * 256 * TFR * 4);
    float*     h2       = (float*)     alloc((size_t)BATCH * 128 * TFR * 4);
    float*     h3       = (float*)     alloc((size_t)BATCH *  64 * TFR * 4);
    float*     pbuf     = (float*)     alloc((size_t)BATCH *  40 * TFR * 4);
    float*     control  = (float*)     alloc((size_t)BATCH *   8 * TFR * 4);
    float*     avgshape = (float*)     alloc((size_t)BATCH * 32 * 4);
    float*     locw     = (float*)     alloc((size_t)BATCH * 5 * 4);
    float*     stdev    = (float*)     alloc((size_t)BATCH * 4);
    float*     fstab    = (float*)     alloc((size_t)5 * 32 * 4);
    float2*    tw1      = (float2*)    alloc((size_t)FN1 * 8);
    float2*    tw2      = (float2*)    alloc((size_t)FN2 * 8);
    float2*    twN      = (float2*)    alloc((size_t)NAUD * 8);
    _Float16*  w1h      = (_Float16*)  alloc((size_t)256 * 672 * 2);
    _Float16*  w2h      = (_Float16*)  alloc((size_t)128 * 768 * 2);
    _Float16*  w3h      = (_Float16*)  alloc((size_t) 64 * 384 * 2);
    _Float16*  w4h      = (_Float16*)  alloc((size_t) 48 *  64 * 2);
    _Float16*  imb      = (_Float16*)  alloc((size_t)NPOS * 768 * 2);  // shared im2col buf
    float*     shaped   = (float*)     alloc((size_t)BATCH * NAUD * 4);
    float2*    bufA     = (float2*)    alloc((size_t)BATCH * NAUD * 8);
    float2*    bufB     = (float2*)    alloc((size_t)BATCH * NAUD * 8);
    (void)in_sizes; (void)n_in; (void)out_size; (void)ws_size;

    const int EL = (BATCH * NAUD + 255) / 256;          // elementwise grid over audio
    const int P1 = BATCH * (FN2 / 8);                   // fft phase1 grid
    const int P2 = BATCH * (FN1 / 8);                   // fft phase2 grid

    init_tables_kernel<<<1, 256, 0, stream>>>(tw1, tw2, twN, fstab, avgshape);
    build_input_kernel<<<(BATCH * 129 * TFR + 255) / 256, 256, 0, stream>>>(cond, f0, xin);

    // ---- pack weights to padded f16 (OC, Ktot) -> (Mpad, Kpad) ----
    pack_weights_kernel<<<(256 * 672 + 255) / 256, 256, 0, stream>>>(w1, w1h, 256, 645, 256, 672);
    pack_weights_kernel<<<(128 * 768 + 255) / 256, 256, 0, stream>>>(w2, w2h, 128, 768, 128, 768);
    pack_weights_kernel<<<( 64 * 384 + 255) / 256, 256, 0, stream>>>(w3, w3h,  64, 384,  64, 384);
    pack_weights_kernel<<<( 48 *  64 + 255) / 256, 256, 0, stream>>>(w4, w4h,  40,  64,  48,  64);

    // ---- conv stack: im2col (f16) + WMMA GEMM per layer ----
    // L1: 129ch k=5 pad=2 -> 256ch   (Kpad=672, Mtiles=16)
    im2col_kernel<<<dim3((672 + 255) / 256, NPOS), 256, 0, stream>>>(xin, imb, 129, 5, 2, 672);
    gemm_wmma_kernel<<<(16 * 800 + 7) / 8, 256, 0, stream>>>(w1h, imb, b1, h1, 672, 256, 1, 16, 800);
    // L2: 256ch k=3 pad=1 -> 128ch   (Kpad=768, Mtiles=8)
    im2col_kernel<<<dim3((768 + 255) / 256, NPOS), 256, 0, stream>>>(h1, imb, 256, 3, 1, 768);
    gemm_wmma_kernel<<<( 8 * 800 + 7) / 8, 256, 0, stream>>>(w2h, imb, b2, h2, 768, 128, 1,  8, 800);
    // L3: 128ch k=3 pad=1 -> 64ch    (Kpad=384, Mtiles=4)
    im2col_kernel<<<dim3((384 + 255) / 256, NPOS), 256, 0, stream>>>(h2, imb, 128, 3, 1, 384);
    gemm_wmma_kernel<<<( 4 * 800 + 7) / 8, 256, 0, stream>>>(w3h, imb, b3, h3, 384,  64, 1,  4, 800);
    // L4: 64ch k=1 pad=0 -> 40ch     (Kpad=64, Mtiles=3)
    im2col_kernel<<<dim3((64 + 255) / 256, NPOS), 256, 0, stream>>>(h3, imb, 64, 1, 0, 64);
    gemm_wmma_kernel<<<( 3 * 800 + 7) / 8, 256, 0, stream>>>(w4h, imb, b4, pbuf, 64,  40, 0,  3, 800);

    activations_kernel<<<(BATCH * TFR + 255) / 256, 256, 0, stream>>>(pbuf, bf_out, control, avgshape);

    // ---- FFT round-trip 1: shaped noise ----
    complexify_kernel<<<EL, 256, 0, stream>>>(wn, bufA);
    fft_phase1_kernel<<<P1, 256, 0, stream>>>(bufA, bufB, tw1, twN, 0);
    fft_phase2_kernel<<<P2, 256, 0, stream>>>(bufB, bufA, tw2, 0);
    apply_gains_kernel<<<EL, 256, 0, stream>>>(bufA, avgshape);
    fft_phase1_kernel<<<P1, 256, 0, stream>>>(bufA, bufB, tw1, twN, 1);
    fft_phase2_kernel<<<P2, 256, 0, stream>>>(bufB, bufA, tw2, 1);
    extract_real_kernel<<<EL, 256, 0, stream>>>(bufA, shaped, 1.f / (float)NAUD);

    std_reduce_kernel<<<BATCH, 256, 0, stream>>>(shaped, stdev);
    locw_reduce_kernel<<<BATCH, 256, 0, stream>>>(control, locw);
    envelope_kernel<<<EL, 256, 0, stream>>>(shaped, stdev, control, xin, bufA);

    // ---- FFT round-trip 2: turbulence filtering ----
    fft_phase1_kernel<<<P1, 256, 0, stream>>>(bufA, bufB, tw1, twN, 0);
    fft_phase2_kernel<<<P2, 256, 0, stream>>>(bufB, bufA, tw2, 0);
    apply_total_filter_kernel<<<EL, 256, 0, stream>>>(bufA, locw, fstab);
    fft_phase1_kernel<<<P1, 256, 0, stream>>>(bufA, bufB, tw1, twN, 1);
    fft_phase2_kernel<<<P2, 256, 0, stream>>>(bufB, bufA, tw2, 1);
    extract_real_kernel<<<EL, 256, 0, stream>>>(bufA, out, 1.f / (float)NAUD);
}
